// SwinTransformerBlock_64381559767607
// MI455X (gfx1250) — compile-verified, matrix operand order checked by reading, not run on hardware
//
#include <hip/hip_runtime.h>

// ---------------------------------------------------------------------------
// Types
// ---------------------------------------------------------------------------
typedef __attribute__((ext_vector_type(16))) _Float16 v16h;
typedef __attribute__((ext_vector_type(8)))  _Float16 v8h;
typedef __attribute__((ext_vector_type(8)))  float    v8f;
typedef __attribute__((ext_vector_type(4)))  int      v4i;

union FragA { v16h v; v8h h[2]; };

// ---------------------------------------------------------------------------
// gfx1250 async global->LDS copies (guarded; fallback = load+ds_store).
// Probe round 1 showed the builtin exists with params (v4i AS1*, v4i AS3*, Ii, Ii).
// ---------------------------------------------------------------------------
#if defined(__HIP_DEVICE_COMPILE__) && defined(__gfx1250__)
#if __has_builtin(__builtin_amdgcn_global_load_async_to_lds_b128)
#define HAVE_ASYNC_LDS 1
#endif
#endif
#ifndef HAVE_ASYNC_LDS
#define HAVE_ASYNC_LDS 0
#endif

#if HAVE_ASYNC_LDS
#define GLOBAL_AS __attribute__((address_space(1)))
#define LDS_AS    __attribute__((address_space(3)))
#if __has_builtin(__builtin_amdgcn_s_wait_asynccnt)
#define ASYNC_WAIT() __builtin_amdgcn_s_wait_asynccnt(0)
#else
#define ASYNC_WAIT() asm volatile("s_wait_asynccnt 0" ::: "memory")
#endif
#else
#define ASYNC_WAIT() do {} while (0)
#endif

// ---------------------------------------------------------------------------
// Constants for this problem size (B=32, H=W=56, C=512, heads=16, ws=7, shift=3)
// ---------------------------------------------------------------------------
#define CC    512
#define HEADS 16
#define HD    32
#define WSZ   7
#define NTOK  49      // 7*7
#define SHF   3
#define IMG   56
#define NWIN  64      // 8*8 windows per image

// ---------------------------------------------------------------------------
// f32 -> f16 conversion (weights)
// ---------------------------------------------------------------------------
__global__ void cvt_f32_to_f16(const float* __restrict__ s,
                               _Float16* __restrict__ d, int n) {
  int i = blockIdx.x * blockDim.x + threadIdx.x;
  int stride = gridDim.x * blockDim.x;
  for (; i < n; i += stride) d[i] = (_Float16)s[i];
}

// ---------------------------------------------------------------------------
// LayerNorm (+ optional windowed gather with cyclic shift) -> f16
// One block (256 threads) per output token; C=512 -> 2 channels/thread.
// windowed=1: token is (window*49 + t) order, source gathered from x with roll(-3).
// ---------------------------------------------------------------------------
__global__ __launch_bounds__(256) void ln_f16_kernel(
    const float* __restrict__ src, const float* __restrict__ gamma,
    const float* __restrict__ beta, _Float16* __restrict__ dst, int windowed) {
  const int token = blockIdx.x;
  size_t srow;
  if (windowed) {
    int w = token / NTOK, t = token - w * NTOK;
    int b = w >> 6, widx = w & (NWIN - 1);
    int nh = widx >> 3, nw = widx & 7;
    int r = t / WSZ, c = t - r * WSZ;
    int gr = nh * WSZ + r + SHF; if (gr >= IMG) gr -= IMG;
    int gc = nw * WSZ + c + SHF; if (gc >= IMG) gc -= IMG;
    srow = ((size_t)b * (IMG * IMG) + gr * IMG + gc) * CC;
  } else {
    srow = (size_t)token * CC;
  }
  const int tid = threadIdx.x;
  float v0 = src[srow + tid];
  float v1 = src[srow + tid + 256];
  float s = v0 + v1;
  float q = v0 * v0 + v1 * v1;
  for (int off = 16; off; off >>= 1) {
    s += __shfl_xor(s, off, 32);
    q += __shfl_xor(q, off, 32);
  }
  __shared__ float sh[16];
  const int wv = tid >> 5;
  if ((tid & 31) == 0) { sh[wv] = s; sh[8 + wv] = q; }
  __syncthreads();
  if (tid < 32) {
    s = (tid < 8) ? sh[tid] : 0.f;
    q = (tid < 8) ? sh[8 + tid] : 0.f;
    for (int off = 4; off; off >>= 1) {
      s += __shfl_xor(s, off, 32);
      q += __shfl_xor(q, off, 32);
    }
    if (tid == 0) { sh[0] = s; sh[8] = q; }
  }
  __syncthreads();
  const float mu  = sh[0] * (1.f / (float)CC);
  const float var = sh[8] * (1.f / (float)CC) - mu * mu;
  const float rs  = rsqrtf(var + 1e-5f);
  const size_t drow = (size_t)token * CC;
  dst[drow + tid]       = (_Float16)((v0 - mu) * rs * gamma[tid] + beta[tid]);
  dst[drow + tid + 256] = (_Float16)((v1 - mu) * rs * gamma[tid + 256] + beta[tid + 256]);
}

// ---------------------------------------------------------------------------
// Tiled WMMA GEMM:  D[M,N] = A[M,K](f16) * W[K,N](f16) + bias, mode-epilogue
//  MODE 0: store f16
//  MODE 1: + gathered residual x, scatter-store f32 to d_out (inverse window map)
//  MODE 2: GELU(exact), store f16
//  MODE 3: d_out += result (natural token order), f32
// Block: 256 threads = 8 waves (4x2), block tile 128x128, K-step 32, dbl-buffer LDS.
// ---------------------------------------------------------------------------
template <int MODE>
__global__ __launch_bounds__(256) void gemm_wmma(
    const _Float16* __restrict__ A, const _Float16* __restrict__ W,
    const float* __restrict__ bias, _Float16* __restrict__ outH,
    float* __restrict__ outF, const float* __restrict__ resid,
    int M, int N, int K) {
  __shared__ _Float16 lsA[2][128][40];   // +8 pad
  __shared__ _Float16 lsB[2][32][136];   // +8 pad
  const int tid  = threadIdx.x;
  const int lane = tid & 31;
  const int lr   = lane & 15, hi = lane >> 4;
  const int wave = tid >> 5;
  const int waveM = wave >> 1, waveN = wave & 1;
  const int blockM = blockIdx.y * 128, blockN = blockIdx.x * 128;

  v8f acc[2][4] = {};

  auto loadTile = [&](int kt, int bufi) {
#pragma unroll
    for (int j = 0; j < 2; ++j) {            // A: 128x32 f16 = 512 b128 chunks
      int chunk = tid * 2 + j;
      int row = chunk >> 2, col = (chunk & 3) * 8;
      const _Float16* gsrc = A + (size_t)(blockM + row) * K + kt + col;
      _Float16* ldst = &lsA[bufi][row][col];
#if HAVE_ASYNC_LDS
      __builtin_amdgcn_global_load_async_to_lds_b128(
          (GLOBAL_AS v4i*)gsrc, (LDS_AS v4i*)ldst, 0, 0);
#else
      *(v8h*)ldst = *(const v8h*)gsrc;
#endif
    }
#pragma unroll
    for (int j = 0; j < 2; ++j) {            // B: 32x128 f16 = 512 b128 chunks
      int chunk = tid * 2 + j;
      int row = chunk >> 4, col = (chunk & 15) * 8;
      const _Float16* gsrc = W + (size_t)(kt + row) * N + blockN + col;
      _Float16* ldst = &lsB[bufi][row][col];
#if HAVE_ASYNC_LDS
      __builtin_amdgcn_global_load_async_to_lds_b128(
          (GLOBAL_AS v4i*)gsrc, (LDS_AS v4i*)ldst, 0, 0);
#else
      *(v8h*)ldst = *(const v8h*)gsrc;
#endif
    }
  };

  const int nk = K >> 5;
  loadTile(0, 0);
  ASYNC_WAIT();
  __syncthreads();

  for (int t = 0; t < nk; ++t) {
    const int bufi = t & 1;
    if (t + 1 < nk) loadTile((t + 1) << 5, bufi ^ 1);

    FragA a[2];
#pragma unroll
    for (int mt = 0; mt < 2; ++mt) {
      int row = waveM * 32 + mt * 16 + lr;
      a[mt].h[0] = *(const v8h*)&lsA[bufi][row][hi * 8];
      a[mt].h[1] = *(const v8h*)&lsA[bufi][row][16 + hi * 8];
    }
#pragma unroll
    for (int nt = 0; nt < 4; ++nt) {
      FragA b;
      const int nc = waveN * 64 + nt * 16 + lr;
#pragma unroll
      for (int i = 0; i < 16; ++i)
        b.v[i] = lsB[bufi][(i >> 3) * 16 + hi * 8 + (i & 7)][nc];
#pragma unroll
      for (int mt = 0; mt < 2; ++mt)
        acc[mt][nt] = __builtin_amdgcn_wmma_f32_16x16x32_f16(
            false, a[mt].v, false, b.v, (short)0, acc[mt][nt], false, false);
    }
    ASYNC_WAIT();
    __syncthreads();
  }

  // epilogue
#pragma unroll
  for (int mt = 0; mt < 2; ++mt) {
#pragma unroll
    for (int nt = 0; nt < 4; ++nt) {
#pragma unroll
      for (int j = 0; j < 8; ++j) {
        const int m = blockM + waveM * 32 + mt * 16 + hi * 8 + j;
        const int n = blockN + waveN * 64 + nt * 16 + lr;
        float v = acc[mt][nt][j] + bias[n];
        if (MODE == 0) {
          outH[(size_t)m * N + n] = (_Float16)v;
        } else if (MODE == 2) {
          v = 0.5f * v * (1.0f + erff(v * 0.70710678118654752f));
          outH[(size_t)m * N + n] = (_Float16)v;
        } else if (MODE == 3) {
          const size_t idx = (size_t)m * N + n;
          outF[idx] = outF[idx] + v;
        } else {  // MODE 1: inverse window scatter + residual
          int w = m / NTOK, tt = m - w * NTOK;
          int b = w >> 6, widx = w & (NWIN - 1);
          int nh = widx >> 3, nw = widx & 7;
          int r = tt / WSZ, c = tt - r * WSZ;
          int gr = nh * WSZ + r + SHF; if (gr >= IMG) gr -= IMG;
          int gc = nw * WSZ + c + SHF; if (gc >= IMG) gc -= IMG;
          const size_t idx = ((size_t)b * (IMG * IMG) + gr * IMG + gc) * CC + n;
          outF[idx] = resid[idx] + v;
        }
      }
    }
  }
}

// ---------------------------------------------------------------------------
// Windowed attention (shifted, masked, rel-pos bias), WMMA 16x16x32.
// Grid: (numWindows, 4); block 128 threads = 4 waves; wave <-> one head.
// ---------------------------------------------------------------------------
__device__ __forceinline__ int mreg(int v) {
  return v < (IMG - WSZ) ? 0 : (v < (IMG - SHF) ? 1 : 2);
}

__global__ __launch_bounds__(128) void attn_wmma(
    const _Float16* __restrict__ qkv, const float* __restrict__ btab,
    _Float16* __restrict__ outw) {
  __shared__ _Float16 sQ[4][64][HD];
  __shared__ _Float16 sK[4][64][HD];
  __shared__ _Float16 sV[4][64][HD];
  __shared__ _Float16 sP[4][16][64];

  const int w    = blockIdx.x;
  const int h0   = blockIdx.y * 4;
  const int tid  = threadIdx.x;
  const int lane = tid & 31;
  const int lr   = lane & 15, hi = lane >> 4;
  const int h    = tid >> 5;  // wave id = head within block

  // cooperative load of q/k/v (3 * 4 heads * 49 rows of 32 f16 -> b128 chunks)
  for (int c = tid; c < 588 * 4; c += 128) {
    int rowid = c >> 2, ch8 = (c & 3) * 8;
    int s = rowid / 196;          // 196 = 4*49
    int rem = rowid - s * 196;
    int hh = rem / NTOK, t = rem - hh * NTOK;
    const v8h* src = (const v8h*)(qkv + ((size_t)w * NTOK + t) * (3 * CC) +
                                  s * CC + (h0 + hh) * HD + ch8);
    _Float16* dp = (s == 0) ? &sQ[hh][t][ch8]
                 : (s == 1) ? &sK[hh][t][ch8] : &sV[hh][t][ch8];
    *(v8h*)dp = *src;
  }
  {  // zero pad rows 49..63
    v8h zz = {};
    for (int c = tid; c < 180 * 4; c += 128) {
      int rowid = c >> 2, ch8 = (c & 3) * 8;
      int s = rowid / 60;         // 60 = 4*15
      int rem = rowid - s * 60;
      int hh = rem / 15, t = NTOK + (rem - hh * 15);
      _Float16* dp = (s == 0) ? &sQ[hh][t][ch8]
                   : (s == 1) ? &sK[hh][t][ch8] : &sV[hh][t][ch8];
      *(v8h*)dp = zz;
    }
  }
  __syncthreads();

  const int widx = w & (NWIN - 1);
  const int nh = widx >> 3, nw = widx & 7;
  const float scale = 0.17677669529663687f;  // 1/sqrt(32)

  for (int qt = 0; qt < 4; ++qt) {
    FragA a;
    {
      const int qrow = qt * 16 + lr;
      a.h[0] = *(const v8h*)&sQ[h][qrow][hi * 8];
      a.h[1] = *(const v8h*)&sQ[h][qrow][16 + hi * 8];
    }
    v8f acc[4] = {};
#pragma unroll
    for (int nt = 0; nt < 4; ++nt) {
      FragA b;
#pragma unroll
      for (int i = 0; i < 16; ++i)
        b.v[i] = sK[h][nt * 16 + lr][(i >> 3) * 16 + hi * 8 + (i & 7)];
      acc[nt] = __builtin_amdgcn_wmma_f32_16x16x32_f16(
          false, a.v, false, b.v, (short)0, acc[nt], false, false);
    }

    // scale + rel-pos bias + shift mask, then row softmax
    float vals[4][8];
#pragma unroll
    for (int nt = 0; nt < 4; ++nt) {
#pragma unroll
      for (int j = 0; j < 8; ++j) {
        const int q = qt * 16 + hi * 8 + j;
        const int n = nt * 16 + lr;
        float sc = -3.0e38f;
        if (q < NTOK && n < NTOK) {
          int qr = q / WSZ, qc = q - qr * WSZ;
          int nr = n / WSZ, nc = n - nr * WSZ;
          int rp = (qr - nr + WSZ - 1) * (2 * WSZ - 1) + (qc - nc + WSZ - 1);
          sc = acc[nt][j] * scale + btab[rp * HEADS + h0 + h];
          int cq = mreg(nh * WSZ + qr) * 3 + mreg(nw * WSZ + qc);
          int cn = mreg(nh * WSZ + nr) * 3 + mreg(nw * WSZ + nc);
          if (cq != cn) sc -= 100.f;
        }
        vals[nt][j] = sc;
      }
    }
#pragma unroll
    for (int j = 0; j < 8; ++j) {
      float mx = fmaxf(fmaxf(vals[0][j], vals[1][j]),
                       fmaxf(vals[2][j], vals[3][j]));
      for (int off = 8; off; off >>= 1) mx = fmaxf(mx, __shfl_xor(mx, off, 32));
      float e0 = __expf(vals[0][j] - mx);
      float e1 = __expf(vals[1][j] - mx);
      float e2 = __expf(vals[2][j] - mx);
      float e3 = __expf(vals[3][j] - mx);
      float sum = e0 + e1 + e2 + e3;
      for (int off = 8; off; off >>= 1) sum += __shfl_xor(sum, off, 32);
      const float inv = 1.0f / sum;
      const int prow = hi * 8 + j;
      sP[h][prow][lr]      = (_Float16)(e0 * inv);
      sP[h][prow][16 + lr] = (_Float16)(e1 * inv);
      sP[h][prow][32 + lr] = (_Float16)(e2 * inv);
      sP[h][prow][48 + lr] = (_Float16)(e3 * inv);
    }

    // out strip = P(16x64) * V(64x32)
    v8f oacc[2] = {};
#pragma unroll
    for (int kc = 0; kc < 2; ++kc) {
      FragA pa;
      pa.h[0] = *(const v8h*)&sP[h][lr][kc * 32 + hi * 8];
      pa.h[1] = *(const v8h*)&sP[h][lr][kc * 32 + 16 + hi * 8];
#pragma unroll
      for (int dt = 0; dt < 2; ++dt) {
        FragA vb;
#pragma unroll
        for (int i = 0; i < 16; ++i)
          vb.v[i] = sV[h][kc * 32 + (i >> 3) * 16 + hi * 8 + (i & 7)][dt * 16 + lr];
        oacc[dt] = __builtin_amdgcn_wmma_f32_16x16x32_f16(
            false, pa.v, false, vb.v, (short)0, oacc[dt], false, false);
      }
    }
#pragma unroll
    for (int dt = 0; dt < 2; ++dt) {
#pragma unroll
      for (int j = 0; j < 8; ++j) {
        const int q = qt * 16 + hi * 8 + j;
        if (q < NTOK)
          outw[((size_t)w * NTOK + q) * CC + (h0 + h) * HD + dt * 16 + lr] =
              (_Float16)oacc[dt][j];
      }
    }
  }
}

// ---------------------------------------------------------------------------
// Host launcher
// ---------------------------------------------------------------------------
extern "C" void kernel_launch(void* const* d_in, const int* in_sizes, int n_in,
                              void* d_out, int out_size, void* d_ws,
                              size_t ws_size, hipStream_t stream) {
  const float* x       = (const float*)d_in[0];
  const float* w_qkv   = (const float*)d_in[1];
  const float* b_qkv   = (const float*)d_in[2];
  const float* w_proj  = (const float*)d_in[3];
  const float* b_proj  = (const float*)d_in[4];
  const float* btab    = (const float*)d_in[5];
  const float* ln1_g   = (const float*)d_in[6];
  const float* ln1_b   = (const float*)d_in[7];
  const float* ln2_g   = (const float*)d_in[8];
  const float* ln2_b   = (const float*)d_in[9];
  const float* w1      = (const float*)d_in[10];
  const float* b1      = (const float*)d_in[11];
  const float* w2      = (const float*)d_in[12];
  const float* b2      = (const float*)d_in[13];
  (void)n_in; (void)out_size; (void)ws_size;

  const int B = in_sizes[0] / (IMG * IMG * CC);
  const int M = B * IMG * IMG;          // 100352 for B=32, multiple of 128

  char* ws = (char*)d_ws;
  size_t off = 0;
  _Float16* wqkv_h  = (_Float16*)(ws + off); off += (size_t)CC * (3 * CC) * 2;
  _Float16* wproj_h = (_Float16*)(ws + off); off += (size_t)CC * CC * 2;
  _Float16* w1_h    = (_Float16*)(ws + off); off += (size_t)CC * (4 * CC) * 2;
  _Float16* w2_h    = (_Float16*)(ws + off); off += (size_t)(4 * CC) * CC * 2;
  _Float16* actH  = (_Float16*)(ws + off); off += (size_t)M * CC * 2;        // LN1 out / LN2 out
  _Float16* big   = (_Float16*)(ws + off); off += (size_t)M * (4 * CC) * 2;  // qkv / ff1
  _Float16* attnw = (_Float16*)(ws + off);                                   // attention out

  float* out = (float*)d_out;

  // 1) convert weights to f16
  cvt_f32_to_f16<<<512, 256, 0, stream>>>(w_qkv, wqkv_h, CC * 3 * CC);
  cvt_f32_to_f16<<<512, 256, 0, stream>>>(w_proj, wproj_h, CC * CC);
  cvt_f32_to_f16<<<512, 256, 0, stream>>>(w1, w1_h, CC * 4 * CC);
  cvt_f32_to_f16<<<512, 256, 0, stream>>>(w2, w2_h, 4 * CC * CC);

  // 2) window-gather (shift) + LN1 -> f16
  ln_f16_kernel<<<M, 256, 0, stream>>>(x, ln1_g, ln1_b, actH, 1);

  // 3) QKV GEMM
  gemm_wmma<0><<<dim3((3 * CC) / 128, M / 128), 256, 0, stream>>>(
      actH, wqkv_h, b_qkv, big, nullptr, nullptr, M, 3 * CC, CC);

  // 4) windowed attention
  attn_wmma<<<dim3(B * NWIN, HEADS / 4), 128, 0, stream>>>(big, btab, attnw);

  // 5) proj GEMM + inverse-window scatter + residual -> d_out (f32)
  gemm_wmma<1><<<dim3(CC / 128, M / 128), 256, 0, stream>>>(
      attnw, wproj_h, b_proj, nullptr, out, x, M, CC, CC);

  // 6) LN2 -> f16
  ln_f16_kernel<<<M, 256, 0, stream>>>(out, ln2_g, ln2_b, actH, 0);

  // 7) MLP up + GELU
  gemm_wmma<2><<<dim3((4 * CC) / 128, M / 128), 256, 0, stream>>>(
      actH, w1_h, b1, big, nullptr, nullptr, M, 4 * CC, CC);

  // 8) MLP down + residual add into d_out
  gemm_wmma<3><<<dim3(CC / 128, M / 128), 256, 0, stream>>>(
      big, w2_h, b2, nullptr, out, nullptr, M, CC, 4 * CC);
}
